// MaxGraphConv_14826227105921
// MI455X (gfx1250) — matrix operand order repaired and checked by reading, not compiled, not run on hardware
//
#include <hip/hip_runtime.h>
#include <hip/hip_bf16.h>
#include <hip/hip_fp16.h>
#include <math.h>

typedef __attribute__((ext_vector_type(16))) _Float16 v16h;
typedef __attribute__((ext_vector_type(8)))  _Float16 v8h;
typedef __attribute__((ext_vector_type(8)))  float    v8f;

#define NB   16
#define NC   256
#define NN_  1024
#define NOUT 512
#define KG   16

union AFrag { v16h v; v8h h[2]; };

// A-matrix fragment (16x32 f16, MxK): lane m=lane&15; lanes 0-15 hold K 0..15,
// lanes 16-31 hold K 16..31 (contiguous, matching B)
__device__ __forceinline__ v16h load_a_frag(const _Float16* rowbase, int ldk, int kk) {
  const int lane = threadIdx.x & 31;
  const int m = lane & 15;
  const int ks = kk + ((lane >> 4) << 4);
  AFrag u;
  u.h[0] = *(const v8h*)(rowbase + (size_t)m * ldk + ks);
  u.h[1] = *(const v8h*)(rowbase + (size_t)m * ldk + ks + 8);
  return u.v;
}

// B-matrix fragment (32x16 f16, KxN) from K-contiguous storage (i.e. B^T row-major):
// lane n=lane&15; lanes 0-15 hold K 0..15, lanes 16-31 hold K 16..31 (contiguous 32B)
__device__ __forceinline__ v16h load_b_frag(const _Float16* rowbase, int ldk, int kk) {
  const int lane = threadIdx.x & 31;
  const int n = lane & 15;
  const int ks = kk + ((lane >> 4) << 4);
  return *(const v16h*)(rowbase + (size_t)n * ldk + ks);
}

// CDNA5 async global->LDS copy (16B per lane), tracked by ASYNCcnt.
// LDS operand = wave-relative byte offset = low 32 bits of the generic shared pointer
// (ISA 10.2: LDS aperture addresses map via addr[31:0]).
__device__ __forceinline__ void async_b128(void* lds, const void* g) {
  const unsigned loff = (unsigned)(size_t)lds;
  asm volatile("global_load_async_to_lds_b128 %0, %1, off"
               :: "v"(loff), "v"(g) : "memory");
}
__device__ __forceinline__ void wait_async0() {
  asm volatile("s_wait_asynccnt 0x0" ::: "memory");
}

// ---------------- kernel 1: convert conv weight to f16 ----------------
__global__ void k_wcvt(const float* __restrict__ w, _Float16* __restrict__ w16) {
  const int i = blockIdx.x * 256 + threadIdx.x;   // 512*512 / 256 blocks
  w16[i] = (_Float16)w[i];
}

// ---------------- kernel 2: L2-normalize rows over C, emit f16 ----------------
__global__ void k_normalize(const float* __restrict__ x, _Float16* __restrict__ xn16) {
  const int gid = blockIdx.x * 256 + threadIdx.x;   // 0..16383
  const int b = gid >> 10, n = gid & 1023;
  const float* xb = x + (size_t)b * NC * NN_ + n;
  float s = 0.0f;
  for (int c = 0; c < NC; ++c) { const float v = xb[(size_t)c * NN_]; s += v * v; }
  const float inv = 1.0f / fmaxf(sqrtf(s), 1e-12f);
  _Float16* o = xn16 + (size_t)gid * NC;
  for (int c = 0; c < NC; ++c) o[c] = (_Float16)(xb[(size_t)c * NN_] * inv);
}

// ---------------- kernel 3: WMMA Gram stripe + LDS top-16 ----------------
// One block per (batch, 16-row tile). A tile (8KB contiguous) is staged into LDS
// with async-to-LDS copies; 4 waves each compute 16 of the 64 16x16 score tiles
// (K=256 -> 8 WMMA each), scores land in a 64KB LDS slab; 16 threads then do
// register-resident top-16 insertion per row.
__global__ void __launch_bounds__(128) k_gram_topk(const _Float16* __restrict__ xn16,
                                                   int* __restrict__ nn_idx) {
  __shared__ float    slab[16 * NN_];    // 64 KB
  __shared__ _Float16 atile[16 * NC];    //  8 KB (of 320 KB WGP LDS)
  const int b  = blockIdx.x >> 6;
  const int rt = blockIdx.x & 63;
  const int rowbase = rt * 16;
  const int wave = threadIdx.x >> 5;
  const int lane = threadIdx.x & 31;
  const _Float16* xb   = xn16 + (size_t)b * NN_ * NC;
  const _Float16* arow = xb + (size_t)rowbase * NC;   // 16*256 f16, fully contiguous

  // stage A tile: 128 threads x 16B x 4 = 8 KB
  {
    const char* src = (const char*)arow;
    char* dst = (char*)atile;
    #pragma unroll
    for (int i = 0; i < 4; ++i) {
      const int off = (int)threadIdx.x * 16 + i * 2048;
      async_b128(dst + off, src + off);
    }
    wait_async0();
  }
  __syncthreads();

  for (int ct = wave; ct < 64; ct += 4) {
    const _Float16* brow = xb + (size_t)(ct * 16) * NC;
    v8f acc = {};
    #pragma unroll
    for (int kk = 0; kk < NC; kk += 32) {
      v16h a  = load_a_frag(atile, NC, kk);    // LDS reads
      v16h bm = load_b_frag(brow, NC, kk);     // streamed global reads
      acc = __builtin_amdgcn_wmma_f32_16x16x32_f16(false, a, false, bm,
                                                   (short)0, acc, false, false);
    }
    const int hi  = lane >> 4;
    const int col = ct * 16 + (lane & 15);
    #pragma unroll
    for (int v = 0; v < 8; ++v)
      slab[(v + 8 * hi) * NN_ + col] = acc[v];   // M = v (+8 for hi half-wave)
  }
  __syncthreads();

  if (threadIdx.x < 16) {
    const int r = threadIdx.x;
    const int selfc = rowbase + r;
    float best[KG]; int bidx[KG];
    #pragma unroll
    for (int i = 0; i < KG; ++i) { best[i] = -1e30f; bidx[i] = 0; }
    const float* row = slab + r * NN_;
    for (int col = 0; col < NN_; ++col) {
      if (col == selfc) continue;            // unit-norm: largest dot = nearest
      const float s = row[col];
      if (s > best[KG - 1]) {
        float cs = s; int ci = col;
        #pragma unroll
        for (int p = 0; p < KG; ++p) {       // fully unrolled -> registers only
          if (cs > best[p]) {
            float tf = best[p]; best[p] = cs; cs = tf;
            int   ti = bidx[p]; bidx[p] = ci; ci = ti;
          }
        }
      }
    }
    int* out = nn_idx + ((size_t)(b * NN_) + selfc) * KG;
    #pragma unroll
    for (int i = 0; i < KG; ++i) out[i] = bidx[i];
  }
}

// ---------------- kernel 4: gather neighbors, build interleaved features ----------------
// feat layout (b, n, 512) so WMMA B-fragments are K-contiguous.
__global__ void k_features(const _Float16* __restrict__ xn16,
                           const int* __restrict__ nn_idx,
                           _Float16* __restrict__ feat) {
  const int b = blockIdx.x >> 10;
  const int n = blockIdx.x & 1023;
  const int c = threadIdx.x;
  const _Float16* xb = xn16 + (size_t)b * NN_ * NC;
  const float xv = (float)xb[(size_t)n * NC + c];
  const int* idx = nn_idx + ((size_t)(b * NN_) + n) * KG;
  float m = 0.0f;
  #pragma unroll
  for (int k = 0; k < KG; ++k) {
    const int j = idx[k];
    m = fmaxf(m, fabsf(xv - (float)xb[(size_t)j * NC + c]));
  }
  _Float16* o = feat + ((size_t)(b * NN_) + n) * NOUT + 2 * c;
  o[0] = (_Float16)xv;   // channel 2c
  o[1] = (_Float16)m;    // channel 2c+1
}

// ---------------- kernel 5: WMMA conv GEMM y = W*feat + bias ----------------
// The 4 waves of a block share (b, mt): the 16x512 W tile (16KB contiguous) is
// staged once per block via async-to-LDS. Each wave: 16x128 output tile
// (8 accumulators), A fragment reused 8x per k-step; B streamed from global.
__global__ void __launch_bounds__(128) k_conv_gemm(const _Float16* __restrict__ w16,
                                                   const _Float16* __restrict__ feat,
                                                   const float* __restrict__ bias,
                                                   float* __restrict__ y) {
  __shared__ _Float16 wtile[16 * NOUT];   // 16 KB
  const int wave = threadIdx.x >> 5;
  const int lane = threadIdx.x & 31;
  const int job  = blockIdx.x * 4 + wave;   // 16 * 32 * 8 = 4096 wave jobs
  const int b    = job >> 8;
  const int rem  = job & 255;
  const int mt   = rem >> 3;                // uniform across the 4 waves of a block
  const int n0   = (rem & 7) * 128;
  const _Float16* gw = w16 + (size_t)(mt * 16) * NOUT;  // 16*512 f16 contiguous
  const _Float16* fb = feat + (size_t)b * NN_ * NOUT;

  // stage W tile: 128 threads x 16B x 8 = 16 KB
  {
    const char* src = (const char*)gw;
    char* dst = (char*)wtile;
    #pragma unroll
    for (int i = 0; i < 8; ++i) {
      const int off = (int)threadIdx.x * 16 + i * 2048;
      async_b128(dst + off, src + off);
    }
    wait_async0();
  }
  __syncthreads();

  v8f acc[8] = {};
  for (int kk = 0; kk < NOUT; kk += 32) {
    v16h a = load_a_frag(wtile, NOUT, kk);   // LDS reads
    #pragma unroll
    for (int s = 0; s < 8; ++s) {
      v16h bf = load_b_frag(fb + (size_t)(n0 + s * 16) * NOUT, NOUT, kk);
      acc[s] = __builtin_amdgcn_wmma_f32_16x16x32_f16(false, a, false, bf,
                                                      (short)0, acc[s], false, false);
    }
  }
  const int hi = lane >> 4;
  const int nl = lane & 15;
  #pragma unroll
  for (int s = 0; s < 8; ++s) {
    #pragma unroll
    for (int v = 0; v < 8; ++v) {
      const int o = mt * 16 + v + 8 * hi;
      const int n = n0 + s * 16 + nl;
      y[((size_t)b * NOUT + o) * NN_ + n] = acc[s][v] + bias[o];
    }
  }
}

// ---------------- kernel 6: per-channel batch stats -> fused scale/shift ----------------
__global__ void k_bnstats(const float* __restrict__ y,
                          const float* __restrict__ gamma,
                          const float* __restrict__ beta,
                          float* __restrict__ scale, float* __restrict__ shift) {
  const int o = blockIdx.x;
  const int t = threadIdx.x;
  float s = 0.0f, q = 0.0f;
  for (int b = 0; b < NB; ++b) {
    const float* p = y + ((size_t)b * NOUT + o) * NN_;
    for (int n = t; n < NN_; n += 256) { const float v = p[n]; s += v; q += v * v; }
  }
  __shared__ float ss[256], sq[256];
  ss[t] = s; sq[t] = q; __syncthreads();
  for (int w = 128; w > 0; w >>= 1) {
    if (t < w) { ss[t] += ss[t + w]; sq[t] += sq[t + w]; }
    __syncthreads();
  }
  if (t == 0) {
    const float invn = 1.0f / (float)(NB * NN_);
    const float mean = ss[0] * invn;
    const float var  = sq[0] * invn - mean * mean;
    const float r    = rsqrtf(var + 1e-5f);
    const float sc   = gamma[o] * r;
    scale[o] = sc;
    shift[o] = beta[o] - mean * sc;
  }
}

// ---------------- kernel 7: in-place BN + exact GELU ----------------
__global__ void k_bngelu(float* __restrict__ y,
                         const float* __restrict__ scale,
                         const float* __restrict__ shift) {
  const size_t i = (size_t)blockIdx.x * 256 + threadIdx.x;
  const int o = (int)((i >> 10) & (NOUT - 1));
  const float v = y[i] * scale[o] + shift[o];
  y[i] = 0.5f * v * (1.0f + erff(v * 0.70710678118654752f));
}

extern "C" void kernel_launch(void* const* d_in, const int* in_sizes, int n_in,
                              void* d_out, int out_size, void* d_ws, size_t ws_size,
                              hipStream_t stream) {
  const float* x     = (const float*)d_in[0];   // (16, 256, 1024)
  const float* w     = (const float*)d_in[1];   // (512, 512)
  const float* bias  = (const float*)d_in[2];   // (512,)
  const float* gamma = (const float*)d_in[3];   // (512,)
  const float* beta  = (const float*)d_in[4];   // (512,)
  float* out = (float*)d_out;                   // (16, 512, 1024)

  // workspace layout (bytes)
  char* ws = (char*)d_ws;
  _Float16* xn16 = (_Float16*)(ws);                          //  8 MiB: (B,N,C) f16
  int*      nn   = (int*)(ws + 8388608);                     //  1 MiB: (B,N,16)
  _Float16* feat = (_Float16*)(ws + 9437184);                // 16 MiB: (B,N,512) f16
  _Float16* w16  = (_Float16*)(ws + 26214400);               // 0.5 MiB
  float*    scl  = (float*)(ws + 26738688);                  // 2 KiB
  float*    shf  = (float*)(ws + 26740736);                  // 2 KiB

  k_wcvt     <<<1024,  256, 0, stream>>>(w, w16);
  k_normalize<<<64,    256, 0, stream>>>(x, xn16);
  k_gram_topk<<<NB*64, 128, 0, stream>>>(xn16, nn);
  k_features <<<NB*NN_,256, 0, stream>>>(xn16, nn, feat);
  k_conv_gemm<<<1024,  128, 0, stream>>>(w16, feat, bias, out);
  k_bnstats  <<<NOUT,  256, 0, stream>>>(out, gamma, beta, scl, shf);
  k_bngelu   <<<32768, 256, 0, stream>>>(out, scl, shf);
}
